// Q4MatMul_91276644974893
// MI455X (gfx1250) — 8-bit matrix operand order repaired and checked by reading, compiled, not run on hardware
//
#include <hip/hip_runtime.h>
#include <hip/hip_bf16.h>

typedef __attribute__((ext_vector_type(8))) int v8i;

#define BH   64
#define MDIM 1024
#define NDIM 1024
#define KDIM 256
#define TILE 128
#define KC   128
#define PITCH 144

__device__ __forceinline__ unsigned int quant_pack4(float a, float b, float c, float d, float inv) {
  float q0 = fminf(fmaxf(rintf(a * inv), -8.0f), 7.0f);
  float q1 = fminf(fmaxf(rintf(b * inv), -8.0f), 7.0f);
  float q2 = fminf(fmaxf(rintf(c * inv), -8.0f), 7.0f);
  float q3 = fminf(fmaxf(rintf(d * inv), -8.0f), 7.0f);
  unsigned int i0 = (unsigned int)(int)q0 & 0xFFu;
  unsigned int i1 = (unsigned int)(int)q1 & 0xFFu;
  unsigned int i2 = (unsigned int)(int)q2 & 0xFFu;
  unsigned int i3 = (unsigned int)(int)q3 & 0xFFu;
  return i0 | (i1 << 8) | (i2 << 16) | (i3 << 24);
}

__global__ __launch_bounds__(256, 1) void q4_wmma_gemm(
    const float* __restrict__ x, const float* __restrict__ y,
    const float* __restrict__ xclip, const float* __restrict__ yclip,
    float* __restrict__ out) {
  __shared__ __align__(16) char As[TILE * PITCH];
  __shared__ __align__(16) char Bs[TILE * PITCH];

  const int t  = threadIdx.x;
  const int bz = blockIdx.z;
  const int m0 = blockIdx.y * TILE;
  const int n0 = blockIdx.x * TILE;

  const float xc = xclip[0];
  const float yc = yclip[0];
  const float invx = 1.0f / xc;
  const float invy = 1.0f / yc;

  const float* xb = x + ((size_t)bz * MDIM + m0) * KDIM;
  const float* yb = y + (size_t)bz * KDIM * NDIM + n0;

  const int wave = t >> 5;
  const int lane = t & 31;
  const int wm   = wave & 3;
  const int wn   = wave >> 2;
  const int h    = lane >> 4;
  const int ln   = lane & 15;

  v8i acc[2][4];
  {
    v8i z = {0, 0, 0, 0, 0, 0, 0, 0};
    #pragma unroll
    for (int i = 0; i < 2; ++i)
      #pragma unroll
      for (int j = 0; j < 4; ++j)
        acc[i][j] = z;
  }

  #pragma unroll
  for (int ko = 0; ko < KDIM / KC; ++ko) {
    if (ko) __syncthreads();

    #pragma unroll
    for (int i = 0; i < 16; ++i) {
      int s   = t + i * 256;
      int row = s >> 5;
      int c4  = s & 31;
      float4 v = *(const float4*)(xb + row * KDIM + ko * KC + c4 * 4);
      *(unsigned int*)(&As[row * PITCH + c4 * 4]) =
          quant_pack4(v.x, v.y, v.z, v.w, invx);
    }

    #pragma unroll
    for (int i = 0; i < 16; ++i) {
      int s  = t + i * 256;
      int kk = s >> 7;
      int n  = s & 127;
      const float* p = yb + (size_t)(ko * KC + kk * 4) * NDIM + n;
      float f0 = p[0];
      float f1 = p[NDIM];
      float f2 = p[2 * NDIM];
      float f3 = p[3 * NDIM];
      *(unsigned int*)(&Bs[n * PITCH + kk * 4]) =
          quant_pack4(f0, f1, f2, f3, invy);
    }

    __syncthreads();

    #pragma unroll
    for (int kc = 0; kc < 2; ++kc) {
      const int kb = kc * 64;

      v8i af[2];
      #pragma unroll
      for (int mt = 0; mt < 2; ++mt) {
        int row = wm * 32 + mt * 16 + ln;
        const char* base = &As[row * PITCH + kb + h * 16];
        int2 d0 = *(const int2*)(base + 0);
        int2 d1 = *(const int2*)(base + 8);
        int2 d2 = *(const int2*)(base + 32);
        int2 d3 = *(const int2*)(base + 40);
        v8i a = {d0.x, d0.y, d1.x, d1.y, d2.x, d2.y, d3.x, d3.y};
        af[mt] = a;
      }

      v8i bf[4];
      #pragma unroll
      for (int nt = 0; nt < 4; ++nt) {
        int col = wn * 64 + nt * 16 + ln;
        const char* base = &Bs[col * PITCH + kb + h * 16];
        int4 q0 = *(const int4*)(base + 0);
        int4 q1 = *(const int4*)(base + 32);
        v8i b = {q0.x, q0.y, q0.z, q0.w, q1.x, q1.y, q1.z, q1.w};
        bf[nt] = b;
      }

      #pragma unroll
      for (int mt = 0; mt < 2; ++mt)
        #pragma unroll
        for (int nt = 0; nt < 4; ++nt)
          acc[mt][nt] = __builtin_amdgcn_wmma_i32_16x16x64_iu8(
              true, af[mt], true, bf[nt], acc[mt][nt], false, false);
    }
  }

  const float scale = xc * yc;
  float* ob = out + ((size_t)bz << 20);
  #pragma unroll
  for (int mt = 0; mt < 2; ++mt) {
    #pragma unroll
    for (int nt = 0; nt < 4; ++nt) {
      int col     = n0 + wn * 64 + nt * 16 + ln;
      int rowbase = m0 + wm * 32 + mt * 16 + h * 8;
      #pragma unroll
      for (int r = 0; r < 8; ++r) {
        ob[(size_t)(rowbase + r) * NDIM + col] = scale * (float)acc[mt][nt][r];
      }
    }
  }
}

extern "C" void kernel_launch(void* const* d_in, const int* in_sizes, int n_in,
                              void* d_out, int out_size, void* d_ws, size_t ws_size,
                              hipStream_t stream) {
  (void)in_sizes; (void)n_in; (void)d_ws; (void)ws_size; (void)out_size;
  const float* x  = (const float*)d_in[0];
  const float* y  = (const float*)d_in[1];
  const float* xc = (const float*)d_in[2];
  const float* yc = (const float*)d_in[3];
  float* out = (float*)d_out;

  dim3 grid(NDIM / TILE, MDIM / TILE, BH);
  q4_wmma_gemm<<<grid, dim3(256), 0, stream>>>(x, y, xc, yc, out);
}
